// SelfAttention_7292854468968
// MI455X (gfx1250) — matrix operand order repaired and checked by reading, not compiled, not run on hardware
//
#include <hip/hip_runtime.h>
#include <hip/hip_bf16.h>
#include <math.h>
#include <stdint.h>

typedef _Float16 half8  __attribute__((ext_vector_type(8)));
typedef _Float16 half16 __attribute__((ext_vector_type(16)));
typedef float    float4v __attribute__((ext_vector_type(4)));
typedef float    float8 __attribute__((ext_vector_type(8)));

#define BATCH 4
#define SEQ   4096
#define DIM   256
#define ROWS  (BATCH * SEQ)          // 16384
#define QK_ELEMS (ROWS * DIM)        // 4,194,304 per matrix (f16)
#define W_ELEMS  (DIM * DIM)         // 65,536 per weight matrix

static __device__ __forceinline__ half16 cat16(half8 lo, half8 hi) {
    half16 r;
#pragma unroll
    for (int i = 0; i < 8; ++i) { r[i] = lo[i]; r[i + 8] = hi[i]; }
    return r;
}

static __device__ __forceinline__ float8 wmma_f16(half16 a, half16 b, float8 c) {
    return __builtin_amdgcn_wmma_f32_16x16x32_f16(false, a, false, b,
                                                  (short)0, c, false, false);
}

// CDNA5 async DMA: per-lane 16B global -> LDS, tracked by ASYNCcnt.
static __device__ __forceinline__ void async_b128(uint32_t lds_off, const void* gaddr) {
    asm volatile("global_load_async_to_lds_b128 %0, %1, off"
                 :: "v"(lds_off), "v"(gaddr) : "memory");
}
static __device__ __forceinline__ void wait_asynccnt0() {
    asm volatile("s_wait_asynccnt 0x0" ::: "memory");
}

// ---------------------------------------------------------------------------
// Prep A: convert x (f32) -> xh (f16), row-major [B*S][256].
// ---------------------------------------------------------------------------
__global__ __launch_bounds__(256)
void convert_x_kernel(const float* __restrict__ x, _Float16* __restrict__ xh)
{
    const int i = (blockIdx.x * 256 + threadIdx.x) * 4;
    float4v v = *(const float4v*)(x + i);
    xh[i + 0] = (_Float16)v.x; xh[i + 1] = (_Float16)v.y;
    xh[i + 2] = (_Float16)v.z; xh[i + 3] = (_Float16)v.w;
}

// ---------------------------------------------------------------------------
// Prep B: transpose+convert Wq/Wk/Wv (f32 [k][n]) -> Wt (f16 [mat][n][k]).
// ---------------------------------------------------------------------------
__global__ __launch_bounds__(256)
void transpose_w_kernel(const float* __restrict__ Wq,
                        const float* __restrict__ Wk,
                        const float* __restrict__ Wv,
                        _Float16* __restrict__ Wt)
{
    const int i   = blockIdx.x * 256 + threadIdx.x;   // 0 .. 3*65536-1
    const int mat = i >> 16;
    const int rem = i & 65535;
    const int n   = rem >> 8;
    const int k   = rem & 255;
    const float* W = (mat == 0) ? Wq : (mat == 1) ? Wk : Wv;
    Wt[i] = (_Float16)W[k * DIM + n];                 // Wt[mat][n][k]
}

// ---------------------------------------------------------------------------
// Kernel 1: QKV projection.  One wave -> four 16x16 tiles (shared A operand).
//   Q, K row-major f16 [B*S][256]; V transposed f16 [B][256][S].
// ---------------------------------------------------------------------------
__global__ __launch_bounds__(128)
void qkv_proj_kernel(const _Float16* __restrict__ xh,
                     const _Float16* __restrict__ Wt,
                     const float* __restrict__ bq,
                     const float* __restrict__ bk,
                     const float* __restrict__ bv,
                     _Float16* __restrict__ Qg,
                     _Float16* __restrict__ Kg,
                     _Float16* __restrict__ Vtg)
{
    const int wave = threadIdx.x >> 5;
    const int lane = threadIdx.x & 31;
    const int id   = blockIdx.x * 4 + wave;          // 0 .. 12287

    const int mat      = id / 4096;                  // 0=Q 1=K 2=V
    const int rem      = id % 4096;
    const int rowTile  = rem >> 2;
    const int colGroup = rem & 3;

    const float* bias = (mat == 0) ? bq : (mat == 1) ? bk : bv;

    const int nl = lane & 15;
    const int hf = lane >> 4;
    const int mrow = rowTile * 16;

    float8 acc[4];
#pragma unroll
    for (int c = 0; c < 4; ++c) acc[c] = (float8){};

    const _Float16* xrow  = xh + (size_t)(mrow + nl) * DIM;
    const _Float16* wbase = Wt + (size_t)mat * W_ELEMS;

#pragma unroll
    for (int k0 = 0; k0 < DIM; k0 += 32) {
        half8 alo = *(const half8*)(xrow + k0 + hf * 16);
        half8 ahi = *(const half8*)(xrow + k0 + hf * 16 + 8);
        half16 a = cat16(alo, ahi);
#pragma unroll
        for (int c = 0; c < 4; ++c) {
            const int col = colGroup * 64 + c * 16 + nl;
            half16 bm = *(const half16*)(wbase + (size_t)col * DIM + k0 + hf * 16);
            acc[c] = wmma_f16(a, bm, acc[c]);
        }
    }

#pragma unroll
    for (int c = 0; c < 4; ++c) {
        const int col  = colGroup * 64 + c * 16 + nl;
        const float bval = bias[col];
#pragma unroll
        for (int r = 0; r < 8; ++r) {
            const float v = acc[c][r] + bval;
            const int grow = mrow + r + 8 * hf;
            if (mat == 0) {
                Qg[(size_t)grow * DIM + col] = (_Float16)v;
            } else if (mat == 1) {
                Kg[(size_t)grow * DIM + col] = (_Float16)v;
            } else {
                const int bb = grow >> 12;
                const int ss = grow & 4095;
                Vtg[((size_t)(bb * DIM + col)) * SEQ + ss] = (_Float16)v;
            }
        }
    }
}

// ---------------------------------------------------------------------------
// Kernel 2: flash attention.  4 waves/block share batch + key range:
// K/V chunks staged once per block into double-buffered LDS via async DMA.
// ---------------------------------------------------------------------------
__global__ __launch_bounds__(128)
void flash_attn_kernel(const _Float16* __restrict__ Qg,
                       const _Float16* __restrict__ Kg,
                       const _Float16* __restrict__ Vtg,
                       float* __restrict__ outp)
{
    __shared__ __align__(32) _Float16 sQ[4][16 * DIM];   // per-wave Q tile
    __shared__ __align__(32) _Float16 sP[4][16 * 32];    // per-wave P tile
    __shared__ __align__(32) _Float16 sK[2][32 * DIM];   // block K chunk [key][dim]
    __shared__ __align__(32) _Float16 sV[2][DIM * 32];   // block V chunk [dim][key]

    const int tid  = threadIdx.x;
    const int wave = tid >> 5;
    const int lane = tid & 31;
    const int idx  = blockIdx.x * 4 + wave;          // 0 .. 1023
    const int bb   = idx >> 8;                       // batch (same for all 4 waves)
    const int qt   = idx & 255;                      // query tile

    const int nl = lane & 15;
    const int hf = lane >> 4;

    // --- stage Q tile (16 x 256 f16) into per-wave LDS ---
    {
        const _Float16* qbase = Qg + ((size_t)(bb * SEQ + qt * 16)) * DIM;
#pragma unroll 4
        for (int i = lane; i < 512; i += 32) {
            const int row = i >> 5;
            const int c8  = (i & 31) * 8;
            *(half8*)&sQ[wave][row * DIM + c8] = *(const half8*)(qbase + (size_t)row * DIM + c8);
        }
    }

    // --- async stage of one 32-key chunk (K: 16KB contiguous, V: 256x64B rows) ---
    auto stage = [&](int j, int buf) {
        const char* kbase = (const char*)(Kg + (size_t)(bb * SEQ + j) * DIM);
        const uint32_t klds = (uint32_t)(uintptr_t)&sK[buf][0];
#pragma unroll
        for (int i = 0; i < 8; ++i) {
            const int c = tid + i * 128;             // 1024 x 16B chunks
            async_b128(klds + c * 16, kbase + c * 16);
        }
        const uint32_t vlds = (uint32_t)(uintptr_t)&sV[buf][0];
#pragma unroll
        for (int i = 0; i < 8; ++i) {
            const int c = tid + i * 128;             // 1024 x 16B chunks
            const int d = c >> 2, part = c & 3;      // dim row, 16B part of 64B row
            const char* g = (const char*)(Vtg + ((size_t)(bb * DIM + d)) * SEQ + j) + part * 16;
            async_b128(vlds + c * 16, g);
        }
    };

    float m_run[8], s_run[8];
#pragma unroll
    for (int r = 0; r < 8; ++r) { m_run[r] = -INFINITY; s_run[r] = 0.0f; }

    float8 acc[16];
#pragma unroll
    for (int nt = 0; nt < 16; ++nt) acc[nt] = (float8){};

    const float scale = 0.0625f;                     // 1/sqrt(256)

    stage(0, 0);                                     // prime the pipeline

    for (int j = 0; j < SEQ; j += 32) {
        const int cur = (j >> 5) & 1;

        wait_asynccnt0();                            // this wave's DMA done
        __syncthreads();                             // all waves' DMA visible; prev compute done
        if (j + 32 < SEQ) stage(j + 32, cur ^ 1);    // overlap next DMA with compute

        // --- scores: two 16x16 tiles (keys j..j+15, j+16..j+31) ---
        float8 s0 = {}, s1 = {};
#pragma unroll
        for (int k0 = 0; k0 < DIM; k0 += 32) {
            half8 qlo = *(const half8*)&sQ[wave][nl * DIM + k0 + hf * 16];
            half8 qhi = *(const half8*)&sQ[wave][nl * DIM + k0 + hf * 16 + 8];
            half16 aq = cat16(qlo, qhi);

            half16 b0 = *(const half16*)&sK[cur][nl * DIM + k0 + hf * 16];
            half16 b1 = *(const half16*)&sK[cur][(16 + nl) * DIM + k0 + hf * 16];
            s0 = wmma_f16(aq, b0, s0);
            s1 = wmma_f16(aq, b1, s1);
        }

        // --- online softmax (row reductions across 16-lane halves) ---
#pragma unroll
        for (int r = 0; r < 8; ++r) {
            float v0 = s0[r] * scale;
            float v1 = s1[r] * scale;
            float mx = fmaxf(v0, v1);
#pragma unroll
            for (int off = 8; off > 0; off >>= 1)
                mx = fmaxf(mx, __shfl_xor(mx, off, 32));
            const float mnew  = fmaxf(m_run[r], mx);
            const float alpha = __expf(m_run[r] - mnew);
            const float p0 = __expf(v0 - mnew);
            const float p1 = __expf(v1 - mnew);
            float rs = p0 + p1;
#pragma unroll
            for (int off = 8; off > 0; off >>= 1)
                rs += __shfl_xor(rs, off, 32);
            s_run[r] = s_run[r] * alpha + rs;
            m_run[r] = mnew;
#pragma unroll
            for (int nt = 0; nt < 16; ++nt) acc[nt][r] *= alpha;
            sP[wave][(r + 8 * hf) * 32 + nl]      = (_Float16)p0;
            sP[wave][(r + 8 * hf) * 32 + 16 + nl] = (_Float16)p1;
        }

        // --- P (16x32) x V (32x256): 16 WMMAs, B from LDS V chunk ---
        half8 plo = *(const half8*)&sP[wave][nl * 32 + hf * 16];
        half8 phi = *(const half8*)&sP[wave][nl * 32 + hf * 16 + 8];
        half16 pa = cat16(plo, phi);

#pragma unroll
        for (int nt = 0; nt < 16; ++nt) {
            half16 bv_ = *(const half16*)&sV[cur][(nt * 16 + nl) * 32 + hf * 16];
            acc[nt] = wmma_f16(pa, bv_, acc[nt]);
        }
    }

    // --- finalize: divide by softmax sum, write f32 output ---
#pragma unroll
    for (int nt = 0; nt < 16; ++nt) {
#pragma unroll
        for (int r = 0; r < 8; ++r) {
            const float v = acc[nt][r] / s_run[r];
            const int grow = qt * 16 + r + 8 * hf;
            outp[((size_t)(bb * SEQ + grow)) * DIM + nt * 16 + nl] = v;
        }
    }
}

// ---------------------------------------------------------------------------
extern "C" void kernel_launch(void* const* d_in, const int* in_sizes, int n_in,
                              void* d_out, int out_size, void* d_ws, size_t ws_size,
                              hipStream_t stream)
{
    const float* x  = (const float*)d_in[0];
    const float* Wq = (const float*)d_in[1];
    const float* bq = (const float*)d_in[2];
    const float* Wk = (const float*)d_in[3];
    const float* bk = (const float*)d_in[4];
    const float* Wv = (const float*)d_in[5];
    const float* bv = (const float*)d_in[6];
    float* out = (float*)d_out;

    // workspace layout (all f16): Q | K | Vt | xh | Wt
    _Float16* Qg  = (_Float16*)d_ws;
    _Float16* Kg  = Qg  + QK_ELEMS;
    _Float16* Vtg = Kg  + QK_ELEMS;
    _Float16* xh  = Vtg + QK_ELEMS;
    _Float16* Wt  = xh  + QK_ELEMS;

    convert_x_kernel<<<dim3(QK_ELEMS / (256 * 4)), dim3(256), 0, stream>>>(x, xh);
    transpose_w_kernel<<<dim3(3 * W_ELEMS / 256), dim3(256), 0, stream>>>(Wq, Wk, Wv, Wt);

    qkv_proj_kernel<<<dim3(12288 / 4), dim3(128), 0, stream>>>(
        xh, Wt, bq, bk, bv, Qg, Kg, Vtg);

    flash_attn_kernel<<<dim3(1024 / 4), dim3(128), 0, stream>>>(
        Qg, Kg, Vtg, out);
}
